// IntraRankAttention_69329362092305
// MI455X (gfx1250) — matrix operand order repaired and checked by reading, not compiled, not run on hardware
//
#include <hip/hip_runtime.h>
#include <hip/hip_bf16.h>
#include <math.h>

// ---------------------------------------------------------------------------
// Types for CDNA5 WMMA (wave32): v_wmma_f32_16x16x32_bf16
// ---------------------------------------------------------------------------
typedef __attribute__((ext_vector_type(16))) __bf16 bf16x16;
typedef __attribute__((ext_vector_type(8)))  float  v8f;

#define D_DIM   128
#define NHEAD   4
#define HIDW    32
#define TILE_ME 32    // edges per block (two 16-row WMMA tiles, B reused)
#define TILE_MO 32    // rows per block in output GEMM
#define LDA_E   392   // 384 + 8 pad (edge kernel A stride, ushorts)
#define LDA_O   136   // 128 + 8 pad (output kernel A stride)

__device__ __forceinline__ unsigned short f32_to_bf16(float f) {
    unsigned u = __float_as_uint(f);
    u += 0x7FFFu + ((u >> 16) & 1u);      // round-to-nearest-even
    return (unsigned short)(u >> 16);
}
__device__ __forceinline__ float bf16_to_f32(unsigned short h) {
    return __uint_as_float(((unsigned)h) << 16);
}
// order-preserving encode of float into uint so atomicMax works for any sign
__device__ __forceinline__ unsigned enc_order(float f) {
    unsigned u = __float_as_uint(f);
    return (u & 0x80000000u) ? ~u : (u | 0x80000000u);
}
__device__ __forceinline__ float dec_order(unsigned e) {
    return (e & 0x80000000u) ? __uint_as_float(e & 0x7FFFFFFFu)
                             : __uint_as_float(~e);
}
#define ENC_NEGINF 0x007FFFFFu   // enc_order(-inf)

__device__ __forceinline__ float gelu_exact(float x) {
    return 0.5f * x * (1.0f + erff(x * 0.70710678118654752f));
}

// A fragment: 16x32 bf16 tile, row-major source (lda in ushorts).
// ISA layout: lane m=lane&15, half=lane>>4; vgpr v<4 -> K=2v+8*half ; v>=4 -> K=16+2(v-4)+8*half
__device__ __forceinline__ bf16x16 load_A_frag(const unsigned short* base, int lda,
                                               int kbase, int lane) {
    int m = lane & 15, half = lane >> 4;
    union { bf16x16 v; unsigned short u[16]; } r;
    const unsigned short* p = base + m * lda + kbase + half * 8;
#pragma unroll
    for (int j = 0; j < 4; ++j) {
        r.u[2 * j]     = p[2 * j];
        r.u[2 * j + 1] = p[2 * j + 1];
        r.u[8 + 2 * j]     = p[16 + 2 * j];
        r.u[8 + 2 * j + 1] = p[16 + 2 * j + 1];
    }
    return r.v;
}

// B fragment from pre-shuffled weights: contiguous 16 ushorts per lane per (ktile,ntile).
// layout offset = (((kt*8 + nt)*32 + lane)*16 + i)
__device__ __forceinline__ bf16x16 load_B_frag(const unsigned short* Wsh,
                                               int kt, int nt, int lane) {
    union { bf16x16 v; uint4 q[2]; } r;
    const uint4* p = (const uint4*)(Wsh + (((size_t)(kt * 8 + nt) * 32 + lane) << 4));
    r.q[0] = p[0];
    r.q[1] = p[1];
    return r.v;
}

__device__ __forceinline__ v8f wmma_bf16(bf16x16 a, bf16x16 b, v8f c) {
    return __builtin_amdgcn_wmma_f32_16x16x32_bf16(
        /*neg_a=*/false, a, /*neg_b=*/false, b,
        /*c_mod=*/(short)0, c, /*reuse_a=*/false, /*reuse_b=*/false);
}

// ---------------------------------------------------------------------------
// Prep: bf16 weight shuffles (B-fragment order), fused bias, buffer inits
// ---------------------------------------------------------------------------
__global__ __launch_bounds__(256) void prep_kernel(
    const float* __restrict__ Wsrc, const float* __restrict__ Wtgt,
    const float* __restrict__ Wbrg, const float* __restrict__ bsrc,
    const float* __restrict__ btgt, const float* __restrict__ bbrg,
    const float* __restrict__ Wmsg, const float* __restrict__ Wout,
    unsigned short* __restrict__ wcat_sh, unsigned short* __restrict__ wmsg_sh,
    unsigned short* __restrict__ wout_sh, float* __restrict__ bhid,
    unsigned* __restrict__ segMax, float* __restrict__ segSum,
    float* __restrict__ grouped, int N)
{
    const long nW1 = 384L * 128, nW2 = 256L * 128, nW3 = 128L * 128;
    const long nSeg = (long)N * NHEAD;
    const long nGrp = (long)N * D_DIM;
    const long total = nW1 + nW2 + nW3 + 128 + nSeg + nSeg + nGrp;
    for (long i = (long)blockIdx.x * blockDim.x + threadIdx.x; i < total;
         i += (long)gridDim.x * blockDim.x) {
        long o = i;
        if (o < nW1) {  // W_cat = [Wsrc;Wtgt;Wbrg] -> B-fragment order
            int j = (int)(o & 15); int lane = (int)(o >> 4) & 31;
            long t = o >> 9; int nt = (int)(t & 7); int kt = (int)(t >> 3);
            int krow = kt * 32 + (lane >> 4) * 8 + (j & 7) + (j >> 3) * 16;
            int c = nt * 16 + (lane & 15);
            float v = (krow < 128) ? Wsrc[krow * 128 + c]
                    : (krow < 256) ? Wtgt[(krow - 128) * 128 + c]
                                   : Wbrg[(krow - 256) * 128 + c];
            wcat_sh[o] = f32_to_bf16(v);
            continue;
        }
        o -= nW1;
        if (o < nW2) {  // W_msg
            int j = (int)(o & 15); int lane = (int)(o >> 4) & 31;
            long t = o >> 9; int nt = (int)(t & 7); int kt = (int)(t >> 3);
            int krow = kt * 32 + (lane >> 4) * 8 + (j & 7) + (j >> 3) * 16;
            int c = nt * 16 + (lane & 15);
            wmsg_sh[o] = f32_to_bf16(Wmsg[krow * 128 + c]);
            continue;
        }
        o -= nW2;
        if (o < nW3) {  // W_out
            int j = (int)(o & 15); int lane = (int)(o >> 4) & 31;
            long t = o >> 9; int nt = (int)(t & 7); int kt = (int)(t >> 3);
            int krow = kt * 32 + (lane >> 4) * 8 + (j & 7) + (j >> 3) * 16;
            int c = nt * 16 + (lane & 15);
            wout_sh[o] = f32_to_bf16(Wout[krow * 128 + c]);
            continue;
        }
        o -= nW3;
        if (o < 128) { bhid[o] = bsrc[o] + btgt[o] + bbrg[o]; continue; }
        o -= 128;
        if (o < nSeg) { segMax[o] = ENC_NEGINF; continue; }
        o -= nSeg;
        if (o < nSeg) { segSum[o] = 0.0f; continue; }
        o -= nSeg;
        grouped[o] = 0.0f;
    }
}

// ---------------------------------------------------------------------------
// K1: 32-edge tiles. hidden GEMM (K=384) -> GELU -> logits; message GEMM (K=256).
// 256 threads = 8 waves; wave w owns output columns [16w,16w+16).
// Each B fragment is reused for both 16-row halves of the tile.
// ---------------------------------------------------------------------------
__global__ __launch_bounds__(256) void edge_attn_kernel(
    const float* __restrict__ cell, const float* __restrict__ brg,
    const int* __restrict__ srcIdx, const int* __restrict__ tgtIdx,
    const unsigned short* __restrict__ wcat_sh, const float* __restrict__ bhid,
    const unsigned short* __restrict__ wmsg_sh, const float* __restrict__ bmsg,
    const float* __restrict__ attnProj,
    float* __restrict__ logits, unsigned short* __restrict__ msgOut,
    unsigned* __restrict__ segMax, int E)
{
    __shared__ unsigned short sA[TILE_ME * LDA_E];
    __shared__ float sLogit[TILE_ME][NHEAD];
    __shared__ int sS[TILE_ME], sT[TILE_ME];

    const int tid = threadIdx.x;
    const int e0 = blockIdx.x * TILE_ME;

    if (tid < TILE_ME) {
        int e = e0 + tid;
        int cl = (e < E) ? e : (E - 1);
        sS[tid] = srcIdx[cl];
        sT[tid] = tgtIdx[cl];
        sLogit[tid][0] = 0.f; sLogit[tid][1] = 0.f;
        sLogit[tid][2] = 0.f; sLogit[tid][3] = 0.f;
    }
    __syncthreads();

    // gather 32 x (src|tgt|brg) rows, f32 -> bf16 into LDS (A matrix, row-major)
    for (int idx = tid; idx < TILE_ME * 384; idx += 256) {
        int r = idx / 384, c = idx - r * 384;
        float v;
        if (c < 128)        v = cell[(long)sS[r] * D_DIM + c];
        else if (c < 256)   v = cell[(long)sT[r] * D_DIM + (c - 128)];
        else {
            int e = e0 + r;
            v = (e < E) ? brg[(long)e * D_DIM + (c - 256)] : 0.f;
        }
        sA[r * LDA_E + c] = f32_to_bf16(v);
    }
    __syncthreads();

    const int wave = tid >> 5, lane = tid & 31;
    const int n = lane & 15, half = lane >> 4;
    const int col = wave * 16 + n;
    const unsigned short* sA_hi = sA + 16 * LDA_E;

    // ---- hidden = [src|tgt|brg] @ W_cat, K = 384 (12 k-steps, 24 WMMAs) ----
    v8f acc0 = {}, acc1 = {};
#pragma unroll
    for (int kt = 0; kt < 12; ++kt) {
        bf16x16 b  = load_B_frag(wcat_sh, kt, wave, lane);
        bf16x16 a0 = load_A_frag(sA,    LDA_E, kt * 32, lane);
        bf16x16 a1 = load_A_frag(sA_hi, LDA_E, kt * 32, lane);
        acc0 = wmma_bf16(a0, b, acc0);
        acc1 = wmma_bf16(a1, b, acc1);
    }
    {
        const float bias = bhid[col];
        const int h = col >> 5, d = col & 31;
        const float ap = attnProj[h * HIDW + d];
#pragma unroll
        for (int r = 0; r < 8; ++r) {
            float x0 = gelu_exact(acc0[r] + bias);
            float x1 = gelu_exact(acc1[r] + bias);
            atomicAdd(&sLogit[r + half * 8][h],      x0 * ap);   // ds_add_f32
            atomicAdd(&sLogit[r + half * 8 + 16][h], x1 * ap);
        }
    }

    // ---- messages = [src|brg] @ W_msg, K = 256 (8 k-steps, 16 WMMAs) ----
    // concat cols: [0,128) -> sA cols [0,128); [128,256) -> sA cols [256,384)
    v8f m0 = {}, m1 = {};
#pragma unroll
    for (int kt = 0; kt < 8; ++kt) {
        int kb = kt * 32;
        int srcCol = (kb < 128) ? kb : (kb + 128);
        bf16x16 b  = load_B_frag(wmsg_sh, kt, wave, lane);
        bf16x16 a0 = load_A_frag(sA,    LDA_E, srcCol, lane);
        bf16x16 a1 = load_A_frag(sA_hi, LDA_E, srcCol, lane);
        m0 = wmma_bf16(a0, b, m0);
        m1 = wmma_bf16(a1, b, m1);
    }
    {
        const float bm = bmsg[col];
#pragma unroll
        for (int r = 0; r < 8; ++r) {
            int row = r + half * 8;
            int ea = e0 + row;
            int eb = e0 + row + 16;
            if (ea < E) msgOut[(long)ea * D_DIM + col] = f32_to_bf16(m0[r] + bm);
            if (eb < E) msgOut[(long)eb * D_DIM + col] = f32_to_bf16(m1[r] + bm);
        }
    }
    __syncthreads();

    // flush logits + running segment max (32 edges x 4 heads = 128 threads)
    if (tid < TILE_ME * NHEAD) {
        int row = tid >> 2, h = tid & 3;
        int e = e0 + row;
        if (e < E) {
            float lg = sLogit[row][h];
            logits[(long)e * NHEAD + h] = lg;
            atomicMax(&segMax[(long)sT[row] * NHEAD + h], enc_order(lg));
        }
    }
}

// ---------------------------------------------------------------------------
// K2: segment sum of exp(l - max) per (tgt, head)
// ---------------------------------------------------------------------------
__global__ __launch_bounds__(256) void seg_sum_kernel(
    const float* __restrict__ logits, const int* __restrict__ tgtIdx,
    const unsigned* __restrict__ segMax, float* __restrict__ segSum, int E)
{
    long idx = (long)blockIdx.x * blockDim.x + threadIdx.x;
    if (idx >= (long)E * NHEAD) return;
    int e = (int)(idx >> 2), h = (int)(idx & 3);
    int t = tgtIdx[e];
    float m = dec_order(segMax[(long)t * NHEAD + h]);
    atomicAdd(&segSum[(long)t * NHEAD + h], __expf(logits[idx] - m));
}

// ---------------------------------------------------------------------------
// K3: score * message scattered into grouped[N,128]
// ---------------------------------------------------------------------------
__global__ __launch_bounds__(256) void aggregate_kernel(
    const unsigned short* __restrict__ msgOut, const float* __restrict__ logits,
    const int* __restrict__ tgtIdx, const unsigned* __restrict__ segMax,
    const float* __restrict__ segSum, float* __restrict__ grouped, int E)
{
    long idx = (long)blockIdx.x * blockDim.x + threadIdx.x;
    if (idx >= (long)E * D_DIM) return;
    int e = (int)(idx >> 7), c = (int)(idx & 127);
    int h = c >> 5;
    int t = tgtIdx[e];
    float m = dec_order(segMax[(long)t * NHEAD + h]);
    float s = segSum[(long)t * NHEAD + h];
    float score = __expf(logits[(long)e * NHEAD + h] - m) / s;
    float mv = bf16_to_f32(msgOut[idx]);
    atomicAdd(&grouped[(long)t * D_DIM + c], mv * score);
}

// ---------------------------------------------------------------------------
// K4: out = grouped @ W_out + b_out  (WMMA, 32-row tiles, B reused per 2 halves)
// ---------------------------------------------------------------------------
__global__ __launch_bounds__(256) void out_gemm_kernel(
    const float* __restrict__ grouped, const unsigned short* __restrict__ wout_sh,
    const float* __restrict__ bout, float* __restrict__ out, int N)
{
    __shared__ unsigned short sA[TILE_MO * LDA_O];
    const int tid = threadIdx.x;
    const int r0 = blockIdx.x * TILE_MO;

    for (int idx = tid; idx < TILE_MO * D_DIM; idx += 256) {
        int r = idx >> 7, c = idx & 127;
        int row = r0 + r;
        float v = (row < N) ? grouped[(long)row * D_DIM + c] : 0.f;
        sA[r * LDA_O + c] = f32_to_bf16(v);
    }
    __syncthreads();

    const int wave = tid >> 5, lane = tid & 31;
    const int n = lane & 15, half = lane >> 4;
    const unsigned short* sA_hi = sA + 16 * LDA_O;
    v8f acc0 = {}, acc1 = {};
#pragma unroll
    for (int kt = 0; kt < 4; ++kt) {
        bf16x16 b  = load_B_frag(wout_sh, kt, wave, lane);
        bf16x16 a0 = load_A_frag(sA,    LDA_O, kt * 32, lane);
        bf16x16 a1 = load_A_frag(sA_hi, LDA_O, kt * 32, lane);
        acc0 = wmma_bf16(a0, b, acc0);
        acc1 = wmma_bf16(a1, b, acc1);
    }
    const int col = wave * 16 + n;
    const float bv = bout[col];
#pragma unroll
    for (int r = 0; r < 8; ++r) {
        int ra = r0 + r + half * 8;
        int rb = ra + 16;
        if (ra < N) out[(long)ra * D_DIM + col] = acc0[r] + bv;
        if (rb < N) out[(long)rb * D_DIM + col] = acc1[r] + bv;
    }
}

// ---------------------------------------------------------------------------
// Host launcher
// ---------------------------------------------------------------------------
extern "C" void kernel_launch(void* const* d_in, const int* in_sizes, int n_in,
                              void* d_out, int out_size, void* d_ws, size_t ws_size,
                              hipStream_t stream) {
    const float* cell  = (const float*)d_in[0];
    const float* brg   = (const float*)d_in[1];
    const int*   eidx  = (const int*)  d_in[2];
    const float* Wsrc  = (const float*)d_in[3];
    const float* bsrc  = (const float*)d_in[4];
    const float* Wtgt  = (const float*)d_in[5];
    const float* btgt  = (const float*)d_in[6];
    const float* Wbrg  = (const float*)d_in[7];
    const float* bbrg  = (const float*)d_in[8];
    const float* attnP = (const float*)d_in[9];
    const float* Wmsg  = (const float*)d_in[10];
    const float* bmsg  = (const float*)d_in[11];
    const float* Wout  = (const float*)d_in[12];
    const float* bout  = (const float*)d_in[13];

    const int N = in_sizes[0] / D_DIM;
    const int E = in_sizes[2] / 2;
    const int* srcI = eidx;
    const int* tgtI = eidx + E;

    char* ws = (char*)d_ws;
    size_t off = 0;
    auto carve = [&](size_t bytes) -> char* {
        char* p = ws + off;
        off = (off + bytes + 255) & ~(size_t)255;
        return p;
    };
    unsigned short* wcat_sh = (unsigned short*)carve(384L * 128 * 2);
    unsigned short* wmsg_sh = (unsigned short*)carve(256L * 128 * 2);
    unsigned short* wout_sh = (unsigned short*)carve(128L * 128 * 2);
    float*          bhid    = (float*)         carve(128 * 4);
    float*          logits  = (float*)         carve((size_t)E * NHEAD * 4);
    unsigned short* msgBuf  = (unsigned short*)carve((size_t)E * D_DIM * 2);
    unsigned*       segMax  = (unsigned*)      carve((size_t)N * NHEAD * 4);
    float*          segSum  = (float*)         carve((size_t)N * NHEAD * 4);
    float*          grouped = (float*)         carve((size_t)N * D_DIM * 4);
    (void)ws_size; (void)n_in; (void)out_size;

    prep_kernel<<<4096, 256, 0, stream>>>(Wsrc, Wtgt, Wbrg, bsrc, btgt, bbrg,
                                          Wmsg, Wout, wcat_sh, wmsg_sh, wout_sh,
                                          bhid, segMax, segSum, grouped, N);

    edge_attn_kernel<<<(E + TILE_ME - 1) / TILE_ME, 256, 0, stream>>>(
        cell, brg, srcI, tgtI, wcat_sh, bhid, wmsg_sh, bmsg, attnP,
        logits, msgBuf, segMax, E);

    {
        long tot = (long)E * NHEAD;
        seg_sum_kernel<<<(unsigned)((tot + 255) / 256), 256, 0, stream>>>(
            logits, tgtI, segMax, segSum, E);
    }
    {
        long tot = (long)E * D_DIM;
        aggregate_kernel<<<(unsigned)((tot + 255) / 256), 256, 0, stream>>>(
            msgBuf, logits, tgtI, segMax, segSum, grouped, E);
    }
    out_gemm_kernel<<<(N + TILE_MO - 1) / TILE_MO, 256, 0, stream>>>(
        grouped, wout_sh, bout, (float*)d_out, N);
}
